// Linear_GCNBody_39376260169852
// MI455X (gfx1250) — matrix operand order repaired and checked by reading, not compiled, not run on hardware
//
#include <hip/hip_runtime.h>
#include <hip/hip_bf16.h>

// ---------------------------------------------------------------------------
// Linear+GCN body for MI455X (gfx1250, wave32).
// Dense 32-wide GEMMs via v_wmma_f32_16x16x32_f16 (one 16-node tile / wave).
// Linear head folded into the encoder pack (score = h @ (W2 w0) + const).
// GCN bias b2@Gw injected through the WMMA C accumulator.
// Edge phase: wave-per-edge, lane=channel, coalesced global_atomic_add_f32
// into L2-resident agg (12.8MB << 192MB L2).
// ---------------------------------------------------------------------------

typedef __attribute__((ext_vector_type(16))) _Float16 v16h;
typedef __attribute__((ext_vector_type(8)))  float    v8f;

#define HID 32
#define WAVES_PER_BLOCK 8

// -------------------- BN batch statistics (lane == channel) -----------------
__global__ __launch_bounds__(256) void bn_stats_kernel(
    const float* __restrict__ x, const float* __restrict__ enc_w1,
    const float* __restrict__ enc_b1, float* __restrict__ bnsum, int N)
{
    int lane  = threadIdx.x & 31;
    int wv    = (blockIdx.x * blockDim.x + threadIdx.x) >> 5;
    int nwave = (gridDim.x * blockDim.x) >> 5;
    float w10 = enc_w1[lane];
    float w11 = enc_w1[HID + lane];
    float b1  = enc_b1[lane];
    float s = 0.f, sq = 0.f;
    for (int n = wv; n < N; n += nwave) {
        int nu = __builtin_amdgcn_readfirstlane(n);      // scalar x loads
        float2 xv = ((const float2*)x)[nu];
        float h = fmaf(xv.x, w10, fmaf(xv.y, w11, b1));
        s += h; sq += h * h;
    }
    atomicAdd(&bnsum[lane], s);
    atomicAdd(&bnsum[HID + lane], sq);
}

// -------------------- degree via edge atomics -------------------------------
__global__ __launch_bounds__(256) void deg_kernel(
    const int* __restrict__ ei, float* __restrict__ deg, int E)
{
    int i = blockIdx.x * blockDim.x + threadIdx.x;
    int stride = gridDim.x * blockDim.x;
    for (int e = i; e < E; e += stride)
        atomicAdd(&deg[ei[E + e]], 1.0f);                // dst row
}

// ------ finalize BN, fold heads, pack f16 weights in WMMA B lane layout -----
__global__ void bn_finalize_kernel(
    const float* __restrict__ bn_gamma, const float* __restrict__ bn_beta,
    const float* __restrict__ enc_w2,   const float* __restrict__ gcn_w,
    const float* __restrict__ enc_b2,   const float* __restrict__ gcn_b,
    const float* __restrict__ w0_w,     const float* __restrict__ wb_w,
    const float* __restrict__ w0_b,     const float* __restrict__ wb_b,
    float* __restrict__ bnsum, float* __restrict__ csc, float* __restrict__ csh,
    float* __restrict__ c2,    float* __restrict__ bg,  float* __restrict__ cbias,
    _Float16* __restrict__ w2h, _Float16* __restrict__ gwh, float invN)
{
    int t = threadIdx.x;                                 // 64 threads
    if (t < HID) {
        float mu  = bnsum[t] * invN;
        float var = bnsum[HID + t] * invN - mu * mu;
        float rs  = rsqrtf(var + 1e-5f);
        float sc  = rs * bn_gamma[t];
        csc[t] = sc;
        csh[t] = bn_beta[t] - mu * sc;
        // c2 = W2 @ w0  (folds linear head before the GEMM)
        float a = 0.f;
        for (int n = 0; n < HID; ++n) a += enc_w2[t * HID + n] * w0_w[n];
        c2[t] = a;
        // bg = b2 @ Gw  (emb-bias contribution to xl, fed via WMMA C operand)
        float b = 0.f;
        for (int k = 0; k < HID; ++k) b += enc_b2[k] * gcn_w[k * HID + t];
        bg[t] = b;
    }
    if (t == 0) {
        float cb = w0_b[0] + wb_b[0];
        for (int c = 0; c < HID; ++c)
            cb += gcn_b[c] * wb_w[c] + enc_b2[c] * w0_w[c];
        cbias[0] = cb;
    }
    // B operand layout (ISA 7.12.2, 16-bit B 32x16): half i of lane L holds
    // K = 8*(L/16) + (i<8 ? i : i+8), column N = (L%16) + 16*halfSel.
    int h = t >> 5, lane = t & 31;
    int kb = (lane >> 4) << 3;
    int n  = (lane & 15) + 16 * h;
    for (int i = 0; i < 16; ++i) {
        int k = kb + (i < 8 ? i : i + 8);
        w2h[(h * 32 + lane) * 16 + i] = (_Float16)enc_w2[k * HID + n];
        gwh[(h * 32 + lane) * 16 + i] = (_Float16)gcn_w[k * HID + n];
    }
}

// -------------------- deg -> deg^{-1/2} in place ----------------------------
__global__ __launch_bounds__(256) void dinv_kernel(float* __restrict__ deg, int N)
{
    int i = blockIdx.x * blockDim.x + threadIdx.x;
    if (i < N) deg[i] = rsqrtf(deg[i] + 1.0f);
}

// ---- main node pipeline: encoder -> emb (WMMA) -> xl (WMMA) + lin scores ---
__global__ __launch_bounds__(256) void node_wmma_kernel(
    const float* __restrict__ x,      const float* __restrict__ enc_w1,
    const float* __restrict__ enc_b1, const float* __restrict__ prelu_a,
    const float* __restrict__ csc,    const float* __restrict__ csh,
    const float* __restrict__ c2,     const float* __restrict__ bg,
    const float* __restrict__ cbias,
    const _Float16* __restrict__ w2h, const _Float16* __restrict__ gwh,
    float* __restrict__ pre, float* __restrict__ xl, int N)
{
    __shared__ float s_w10[HID], s_w11[HID], s_b1[HID];
    __shared__ float s_sc[HID], s_sh[HID], s_c2[HID];
    __shared__ float s_tr[WAVES_PER_BLOCK][16 * HID];    // emb tile transpose

    int tid = threadIdx.x;
    if (tid < HID) {
        s_w10[tid] = enc_w1[tid];
        s_w11[tid] = enc_w1[HID + tid];
        s_b1[tid]  = enc_b1[tid];
        s_sc[tid]  = csc[tid];
        s_sh[tid]  = csh[tid];
        s_c2[tid]  = c2[tid];
    }
    __syncthreads();

    int lane = tid & 31;
    int wv   = tid >> 5;
    int nb   = (blockIdx.x * WAVES_PER_BLOCK + wv) * 16; // 16-node tile base
    bool valid = (nb < N);
    if (!valid) nb = 0;

    int mrow  = lane & 15;          // A-layout: row M = lane%16
    int hi    = lane >> 4;          // lane half selects K-range
    int kbase = hi * 8;             // A K map: k = kbase + (i<8 ? i : i+8)
    float slope = prelu_a[0];

    // --- fused encoder front-end: (x @ W1 + b1) -> BN -> PReLU -> f16 A tile
    //     with linear-head partial (h . c2) accumulated in the same pass.
    int nodeA = nb + mrow; if (nodeA >= N) nodeA = N - 1;
    float2 xv = ((const float2*)x)[nodeA];
    v16h aenc;
    float part = 0.f;
#pragma unroll
    for (int i = 0; i < 16; ++i) {
        int k = kbase + (i < 8 ? i : i + 8);
        float h = fmaf(xv.x, s_w10[k], fmaf(xv.y, s_w11[k], s_b1[k]));
        h = fmaf(h, s_sc[k], s_sh[k]);
        h = h > 0.0f ? h : slope * h;
        aenc[i] = (_Float16)h;
        part = fmaf(h, s_c2[k], part);
    }

    // combine the two complementary K-halves (lanes L and L+16) and store
    float p = part + __shfl_xor(part, 16);
    {
        int node = nb + mrow;
        if (valid && hi == 0 && node < N) pre[node] = p + cbias[0];
    }

    // --- emb (minus b2) = A x enc_w2 (two 16-col halves), f32 accum
    const v16h b20 = *(const v16h*)(w2h + (0 * 32 + lane) * 16);
    const v16h b21 = *(const v16h*)(w2h + (1 * 32 + lane) * 16);
    v8f c0 = {}, c1 = {};
    c0 = __builtin_amdgcn_wmma_f32_16x16x32_f16(false, aenc, false, b20,
                                                (short)0, c0, false, false);
    c1 = __builtin_amdgcn_wmma_f32_16x16x32_f16(false, aenc, false, b21,
                                                (short)0, c1, false, false);

    // --- transpose emb tile (C layout -> A layout) through LDS
    // C layout: reg r, lane -> (M = r + 8*hi, N = mrow [+16 for c1])
    float* T = s_tr[wv];
#pragma unroll
    for (int r = 0; r < 8; ++r) {
        T[(r + hi * 8) * HID + mrow]      = c0[r];
        T[(r + hi * 8) * HID + 16 + mrow] = c1[r];
    }
    __syncthreads();

    const float* Trow = &s_tr[wv][mrow * HID];
    v16h aemb;
#pragma unroll
    for (int i = 0; i < 16; ++i) {
        int k = kbase + (i < 8 ? i : i + 8);
        aemb[i] = (_Float16)Trow[k];
    }

    // --- xl = emb @ gcn_w, with C preloaded to bg = b2@Gw (bias for free)
    float bgn0 = bg[mrow], bgn1 = bg[16 + mrow];
    v8f d0, d1;
#pragma unroll
    for (int r = 0; r < 8; ++r) { d0[r] = bgn0; d1[r] = bgn1; }
    const v16h g0 = *(const v16h*)(gwh + (0 * 32 + lane) * 16);
    const v16h g1 = *(const v16h*)(gwh + (1 * 32 + lane) * 16);
    d0 = __builtin_amdgcn_wmma_f32_16x16x32_f16(false, aemb, false, g0,
                                                (short)0, d0, false, false);
    d1 = __builtin_amdgcn_wmma_f32_16x16x32_f16(false, aemb, false, g1,
                                                (short)0, d1, false, false);

    if (valid) {
#pragma unroll
        for (int r = 0; r < 8; ++r) {
            int node = nb + r + hi * 8;
            if (node < N) {
                xl[node * HID + mrow]      = d0[r];
                xl[node * HID + 16 + mrow] = d1[r];
            }
        }
    }
}

// -------- edge scatter: wave per edge, lane == channel, f32 atomics ---------
__global__ __launch_bounds__(256) void scatter_kernel(
    const int* __restrict__ ei, const float* __restrict__ dinv,
    const float* __restrict__ xl, float* __restrict__ agg, int E)
{
    int lane  = threadIdx.x & 31;
    int wv    = (blockIdx.x * blockDim.x + threadIdx.x) >> 5;
    int nwave = (gridDim.x * blockDim.x) >> 5;
    for (int e = wv; e < E; e += nwave) {
        int eu = __builtin_amdgcn_readfirstlane(e);      // scalar edge fetch
        int s = ei[eu];
        int d = ei[E + eu];
        float coef = dinv[s] * dinv[d];
        float v = xl[s * HID + lane] * coef;             // coalesced 128B gather
        atomicAdd(&agg[d * HID + lane], v);              // coalesced 128B atomics
    }
}

// -------- finalize: out = pre + ((agg + xl*dinv^2) . wb_w) ------------------
__global__ __launch_bounds__(256) void finalize_kernel(
    const float* __restrict__ pre, const float* __restrict__ xl,
    const float* __restrict__ agg, const float* __restrict__ dinv,
    const float* __restrict__ wb_w, float* __restrict__ out, int N)
{
    int i = blockIdx.x * blockDim.x + threadIdx.x;
    if (i >= N) return;
    float dv = dinv[i];
    float d2 = dv * dv;
    const float4* xa = (const float4*)(xl + (size_t)i * HID);
    const float4* ag = (const float4*)(agg + (size_t)i * HID);
    const float4* wb = (const float4*)wb_w;
    float acc = 0.f;
#pragma unroll
    for (int q = 0; q < 8; ++q) {
        float4 xq = xa[q], aq = ag[q], wq = wb[q];
        acc += (aq.x + xq.x * d2) * wq.x + (aq.y + xq.y * d2) * wq.y +
               (aq.z + xq.z * d2) * wq.z + (aq.w + xq.w * d2) * wq.w;
    }
    out[i] = pre[i] + acc;
}

// ---------------------------------------------------------------------------
extern "C" void kernel_launch(void* const* d_in, const int* in_sizes, int n_in,
                              void* d_out, int out_size, void* d_ws, size_t ws_size,
                              hipStream_t stream)
{
    const float* x        = (const float*)d_in[0];
    const int*   ei       = (const int*)  d_in[1];
    const float* enc_w1   = (const float*)d_in[2];
    const float* enc_b1   = (const float*)d_in[3];
    const float* bn_gamma = (const float*)d_in[4];
    const float* bn_beta  = (const float*)d_in[5];
    const float* prelu_a  = (const float*)d_in[6];
    const float* enc_w2   = (const float*)d_in[7];
    const float* enc_b2   = (const float*)d_in[8];
    const float* w0_w     = (const float*)d_in[9];
    const float* w0_b     = (const float*)d_in[10];
    const float* gcn_w    = (const float*)d_in[11];
    const float* gcn_b    = (const float*)d_in[12];
    const float* wb_w     = (const float*)d_in[13];
    const float* wb_b     = (const float*)d_in[14];
    float* out = (float*)d_out;

    int N = in_sizes[0] / 2;
    int E = in_sizes[1] / 2;
    size_t Na = (size_t)((N + 31) & ~31);

    float* w      = (float*)d_ws;
    float* bnsum  = w;                    // 64 f32
    float* csc    = w + 64;               // 32
    float* csh    = w + 96;               // 32
    float* c2     = w + 128;              // 32  (W2 @ w0)
    float* bg     = w + 160;              // 32  (b2 @ Gw)
    float* cbias  = w + 192;              // 1 (pad to 224)
    _Float16* w2h = (_Float16*)(w + 224); // 1024 halves
    _Float16* gwh = (_Float16*)(w + 736); // 1024 halves
    float* deg    = w + 1248;             // N  (becomes dinv in place)
    float* pre    = deg + Na;             // N
    float* xl     = pre + Na;             // 32N
    float* agg    = xl + 32 * Na;         // 32N

    // zero the accumulated regions (d_ws is poisoned)
    hipMemsetAsync(bnsum, 0, 64 * sizeof(float), stream);
    hipMemsetAsync(deg, 0, (size_t)N * sizeof(float), stream);
    hipMemsetAsync(agg, 0, 32ull * (size_t)N * sizeof(float), stream);

    bn_stats_kernel<<<512, 256, 0, stream>>>(x, enc_w1, enc_b1, bnsum, N);
    int degGrid = (E + 255) / 256; if (degGrid > 4096) degGrid = 4096;
    deg_kernel<<<degGrid, 256, 0, stream>>>(ei, deg, E);
    bn_finalize_kernel<<<1, 64, 0, stream>>>(bn_gamma, bn_beta, enc_w2, gcn_w,
                                             enc_b2, gcn_b, w0_w, wb_w, w0_b, wb_b,
                                             bnsum, csc, csh, c2, bg, cbias,
                                             w2h, gwh, 1.0f / (float)N);
    dinv_kernel<<<(N + 255) / 256, 256, 0, stream>>>(deg, N);

    int nTiles  = (N + 15) / 16;
    int nBlocks = (nTiles + WAVES_PER_BLOCK - 1) / WAVES_PER_BLOCK;
    node_wmma_kernel<<<nBlocks, 256, 0, stream>>>(
        x, enc_w1, enc_b1, prelu_a, csc, csh, c2, bg, cbias,
        w2h, gwh, pre, xl, N);

    scatter_kernel<<<4096, 256, 0, stream>>>(ei, deg /*now dinv*/, xl, agg, E);

    finalize_kernel<<<(N + 255) / 256, 256, 0, stream>>>(
        pre, xl, agg, deg /*dinv*/, wb_w, out, N);

    (void)n_in; (void)out_size; (void)ws_size;
}
